// SwitchRouter_14998025797841
// MI455X (gfx1250) — matrix operand order repaired and checked by reading, not compiled, not run on hardware
//
#include <hip/hip_runtime.h>
#include <hip/hip_bf16.h>

#define D_MODEL 4096
#define N_EXP   64
#define N_TOK   16384
#define KC      128        // K-chunk staged in LDS
#define ROWS    128        // tokens per block
#define NCHUNK  128        // token chunks for ranking (128 tokens each)
#define CAPACITY 320       // int(1.25 * 16384 / 64)

typedef __attribute__((ext_vector_type(16))) __bf16       v16bf;
typedef __attribute__((ext_vector_type(8)))  float        v8f;
typedef __attribute__((ext_vector_type(4)))  unsigned int u32x4;
typedef __attribute__((ext_vector_type(4)))  float        f32x4;

union Frag16 { u32x4 u[2]; v16bf v; };   // 8 VGPRs of bf16 WMMA operand

// LDS layout (bytes):
//   phase 1:  lA  [128][136] bf16 = 34816   lBT [64][136] bf16 = 17408  (total 52224)
//   phase 2:  lLog[8][16][65] f32 = 33280   (aliases the same buffer)
#define LDS_BYTES 52224
#define LA_STRIDE 136      // KC + 8 bf16 pad -> bank-conflict-free b128 reads
#define LOG_STRIDE 65      // 64 + 1 pad

__global__ __launch_bounds__(256, 1)
void router_gemm_softmax(const float* __restrict__ hidden,
                         const float* __restrict__ gate_w,
                         float* __restrict__ probs_out,
                         float* __restrict__ idxf_out,
                         int*   __restrict__ idx_ws) {
    __shared__ __align__(16) unsigned char smem[LDS_BYTES];
    __bf16* lA   = (__bf16*)smem;                  // [128][LA_STRIDE]
    __bf16* lBT  = (__bf16*)(smem + 34816);        // [64][LA_STRIDE] (B transposed)
    float*  lLog = (float*)smem;                   // [8][16][LOG_STRIDE]

    const int tid  = threadIdx.x;
    const int lane = tid & 31;
    const int wave = tid >> 5;
    const int rowBlock = blockIdx.x * ROWS;

    v8f acc[4];
#pragma unroll
    for (int nt = 0; nt < 4; ++nt)
#pragma unroll
        for (int r = 0; r < 8; ++r) acc[nt][r] = 0.0f;

    // per-lane WMMA fragment addressing (ISA 16-bit A 16x32 / B 32x16 layouts)
    const int rowL    = wave * 16 + (lane & 15);   // A row this lane feeds
    const int kb      = (lane >> 4) * 8;           // A: lanes 16-31 hold K+8 / K+24
    const int ko      = (lane >> 4) * 8;           // B: lanes 16-31 hold K+8 / K+24
    const int colBase = lane & 15;

    for (int ck = 0; ck < D_MODEL / KC; ++ck) {
        __syncthreads();
        // ---- stage A chunk: 128 rows x 128 cols fp32 -> bf16 in LDS ----
        const float* gA = hidden + (size_t)rowBlock * D_MODEL + ck * KC;
#pragma unroll
        for (int it = 0; it < 16; ++it) {
            int i  = tid + it * 256;               // 0..4095 float4 slots
            int r  = i >> 5;
            int c4 = i & 31;
            f32x4 v = *(const f32x4*)(gA + (size_t)r * D_MODEL + c4 * 4);
            __bf16* dst = lA + r * LA_STRIDE + c4 * 4;
            dst[0] = (__bf16)v.x; dst[1] = (__bf16)v.y;
            dst[2] = (__bf16)v.z; dst[3] = (__bf16)v.w;
        }
        // ---- stage B chunk transposed: gate_w[k][e] -> lBT[e][k] ----
        const float* gB = gate_w + (size_t)(ck * KC) * N_EXP;
#pragma unroll
        for (int it = 0; it < 8; ++it) {
            int i  = tid + it * 256;               // 0..2047 float4 slots
            int kr = i >> 4;
            int e4 = i & 15;
            f32x4 v = *(const f32x4*)(gB + (size_t)kr * N_EXP + e4 * 4);
            lBT[(e4 * 4 + 0) * LA_STRIDE + kr] = (__bf16)v.x;
            lBT[(e4 * 4 + 1) * LA_STRIDE + kr] = (__bf16)v.y;
            lBT[(e4 * 4 + 2) * LA_STRIDE + kr] = (__bf16)v.z;
            lBT[(e4 * 4 + 3) * LA_STRIDE + kr] = (__bf16)v.w;
        }
        __syncthreads();

        // prefetch the entire next A chunk (128 rows x 512B = 512 cachelines-ish;
        // 256 threads x 2 prefetches, one per 128B line) -> global_prefetch_b8
        if (ck + 1 < D_MODEL / KC) {
            const float* nf = gA + KC;
#pragma unroll
            for (int p = 0; p < 2; ++p) {
                int slot = tid + p * 256;          // 0..511
                int pr   = slot >> 2;              // row 0..127
                int pc   = (slot & 3) * 32;        // 4 x 128B lines per row-chunk
                __builtin_prefetch(nf + (size_t)pr * D_MODEL + pc, 0, 1);
            }
        }

        // ---- 16 WMMAs per wave per chunk; preload all frags per K-step so the
        //      10 ds_load_b128 overlap and the 4 WMMAs issue back-to-back ----
#pragma unroll
        for (int ks = 0; ks < KC / 32; ++ks) {
            const int k0 = ks * 32;
            Frag16 a;
            const __bf16* ap = lA + rowL * LA_STRIDE + k0 + kb;
            a.u[0] = *(const u32x4*)ap;
            a.u[1] = *(const u32x4*)(ap + 16);
            Frag16 b[4];
#pragma unroll
            for (int nt = 0; nt < 4; ++nt) {
                const __bf16* bp = lBT + (nt * 16 + colBase) * LA_STRIDE + k0 + ko;
                b[nt].u[0] = *(const u32x4*)bp;
                b[nt].u[1] = *(const u32x4*)(bp + 16);
            }
#pragma unroll
            for (int nt = 0; nt < 4; ++nt)
                acc[nt] = __builtin_amdgcn_wmma_f32_16x16x32_bf16(
                    false, a.v, false, b[nt].v, (short)0, acc[nt], false, false);
        }
    }

    __syncthreads();   // all waves done reading lA/lBT; reuse LDS for logits

    // park logits: C/D layout -> lanes 0-15: M=r, lanes 16-31: M=r+8
    const int hl = lane >> 4;
#pragma unroll
    for (int nt = 0; nt < 4; ++nt)
#pragma unroll
        for (int r = 0; r < 8; ++r) {
            int M = r + 8 * hl;
            lLog[(wave * 16 + M) * LOG_STRIDE + nt * 16 + (lane & 15)] = acc[nt][r];
        }
    __syncthreads();

    // softmax + argmax: lanes 0-15 each own one token row
    if (lane < 16) {
        const int t = rowBlock + wave * 16 + lane;
        const float* row = lLog + (wave * 16 + lane) * LOG_STRIDE;
        float mx = row[0];
        int   am = 0;
#pragma unroll
        for (int e = 1; e < N_EXP; ++e) {
            float v = row[e];
            if (v > mx) { mx = v; am = e; }
        }
        float sum = 0.0f;
#pragma unroll
        for (int e = 0; e < N_EXP; ++e) sum += __expf(row[e] - mx);
        const float inv = 1.0f / sum;
        float* po = probs_out + (size_t)t * N_EXP;
#pragma unroll
        for (int q = 0; q < N_EXP / 4; ++q) {
            f32x4 v;
            v.x = __expf(row[q * 4 + 0] - mx) * inv;
            v.y = __expf(row[q * 4 + 1] - mx) * inv;
            v.z = __expf(row[q * 4 + 2] - mx) * inv;
            v.w = __expf(row[q * 4 + 3] - mx) * inv;
            *(f32x4*)(po + q * 4) = v;
        }
        idxf_out[t] = (float)am;
        idx_ws[t]   = am;
    }
}

// ---- phase 2: per-chunk expert histograms ----
__global__ void hist_kernel(const int* __restrict__ idx, int* __restrict__ hist) {
    __shared__ int h[N_EXP];
    h[threadIdx.x] = 0;
    __syncthreads();
    const int c = blockIdx.x;
    for (int j = threadIdx.x; j < (N_TOK / NCHUNK); j += N_EXP)
        atomicAdd(&h[idx[c * (N_TOK / NCHUNK) + j]], 1);
    __syncthreads();
    hist[c * N_EXP + threadIdx.x] = h[threadIdx.x];
}

// ---- phase 3: exclusive scan over chunks per expert + overflow count ----
__global__ void scan_kernel(const int* __restrict__ hist, int* __restrict__ prefix,
                            float* __restrict__ overflow_out) {
    const int e = threadIdx.x;   // one thread per expert
    int run = 0;
    for (int c = 0; c < NCHUNK; ++c) {
        prefix[c * N_EXP + e] = run;
        run += hist[c * N_EXP + e];
    }
    __shared__ int red[N_EXP];
    red[e] = (run > CAPACITY) ? (run - CAPACITY) : 0;
    __syncthreads();
    for (int s = N_EXP / 2; s > 0; s >>= 1) {
        if (e < s) red[e] += red[e + s];
        __syncthreads();
    }
    if (e == 0) *overflow_out = (float)red[0];
}

// ---- phase 4: in-order rank within chunk -> dispatch mask ----
__global__ void rank_kernel(const int* __restrict__ idx, const int* __restrict__ prefix,
                            float* __restrict__ mask_out) {
    const int c = blockIdx.x;
    const int e = threadIdx.x;
    int cnt = prefix[c * N_EXP + e];
    for (int j = 0; j < (N_TOK / NCHUNK); ++j) {
        const int t  = c * (N_TOK / NCHUNK) + j;
        if (idx[t] == e) {
            ++cnt;
            mask_out[t] = (cnt <= CAPACITY) ? 1.0f : 0.0f;
        }
    }
}

extern "C" void kernel_launch(void* const* d_in, const int* in_sizes, int n_in,
                              void* d_out, int out_size, void* d_ws, size_t ws_size,
                              hipStream_t stream) {
    const float* hidden = (const float*)d_in[0];   // (4,4096,4096) fp32
    const float* gate_w = (const float*)d_in[1];   // (4096,64) fp32

    float* out   = (float*)d_out;
    float* probs = out;                                    // N*E floats
    float* idxf  = out + (size_t)N_TOK * N_EXP;            // N floats
    float* mask  = idxf + N_TOK;                           // N floats
    float* ovf   = mask + N_TOK;                           // 1 float

    int* wsIdx  = (int*)d_ws;                              // N ints
    int* hist   = wsIdx + N_TOK;                           // NCHUNK*64 ints
    int* prefix = hist + NCHUNK * N_EXP;                   // NCHUNK*64 ints

    router_gemm_softmax<<<dim3(N_TOK / ROWS), dim3(256), 0, stream>>>(
        hidden, gate_w, probs, idxf, wsIdx);
    hist_kernel<<<dim3(NCHUNK), dim3(N_EXP), 0, stream>>>(wsIdx, hist);
    scan_kernel<<<dim3(1), dim3(N_EXP), 0, stream>>>(hist, prefix, ovf);
    rank_kernel<<<dim3(NCHUNK), dim3(N_EXP), 0, stream>>>(wsIdx, prefix, mask);
}
